// Flashback_55937654063570
// MI455X (gfx1250) — matrix operand order repaired and checked by reading, not compiled, not run on hardware
//
#include <hip/hip_runtime.h>
#include <hip/hip_bf16.h>

typedef _Float16 half_t;
typedef __attribute__((ext_vector_type(16))) _Float16 v16h;
typedef __attribute__((ext_vector_type(8)))  float    v8f;

#define T_LEN 256
#define U_LEN 512
#define H_DIM 64
#define P_DIM 1024
#define G3    192

// A-operand K index for 16-bit WMMA (16x32 MxK):
// lanes 0-15: K = {0..7, 16..23}; lanes 16-31: K = {8..15, 24..31}
__device__ __forceinline__ int a_k(int lane, int e) {
  int kb = (lane & 16) ? 8 : 0;
  return kb + e + ((e < 8) ? 0 : 8);
}
// B-operand K (row) index for 16-bit WMMA (32x16 KxN):
// lanes 0-15: K = {0..7, 16..23}; lanes 16-31: K = {8..15, 24..31}
__device__ __forceinline__ int b_k(int lane, int e) {
  return ((lane & 16) ? 8 : 0) + e + ((e < 8) ? 0 : 8);
}

__device__ __forceinline__ v8f wmma16(v16h a, v16h b, v8f c) {
  return __builtin_amdgcn_wmma_f32_16x16x32_f16(
      false, a, false, b, (short)0, c, false, false);
}
__device__ __forceinline__ float sigm(float x) { return 1.0f / (1.0f + __expf(-x)); }

struct alignas(16) h8pack { half_t v[8]; };

// ---------------------------------------------------------------------------
// Kernel 0: convert weights to f16 *pre-swizzled into WMMA B-operand layout*
// (inner loops load B as one aligned v16h), convert poi_emb to f16 (plain
// layout: A-operand runs are contiguous 16B), gather user embeddings.
//   packed index: (((ks*NT + nt)*32 + lane)*16 + e)  <-  W[k, col]
//   with k = ks*32 + b_k(lane,e),  col = nt*16 + (lane&15)
// ---------------------------------------------------------------------------
__global__ void prep_kernel(const float* __restrict__ Wx, const float* __restrict__ Wh,
                            const float* __restrict__ fcW, const int* __restrict__ au,
                            const float* __restrict__ uemb, const float* __restrict__ poi,
                            half_t* __restrict__ wxp, half_t* __restrict__ whp,
                            half_t* __restrict__ fcwp, half_t* __restrict__ puh,
                            half_t* __restrict__ poih) {
  int i = blockIdx.x * blockDim.x + threadIdx.x;
  const int n0 = 64 * G3;                 // Wx packed (2 ks * 12 nt * 32 * 16)
  const int n1 = n0 + 64 * G3;            // Wh packed
  const int n2 = n1 + 128 * P_DIM;        // fcW packed (4 ks * 64 nt * 32 * 16)
  const int n3 = n2 + U_LEN * H_DIM;      // p_u gather
  const int n4 = n3 + P_DIM * H_DIM;      // poi_emb f16 copy
  if (i < n1) {                           // Wx or Wh
    int j = (i < n0) ? i : (i - n0);
    int e = j & 15, lane = (j >> 4) & 31, rest = j >> 9;
    int nt = rest % 12, ks = rest / 12;
    int k = ks * 32 + b_k(lane, e);
    int col = nt * 16 + (lane & 15);
    if (i < n0) wxp[j] = (half_t)Wx[k * G3 + col];
    else        whp[j] = (half_t)Wh[k * G3 + col];
  } else if (i < n2) {
    int j = i - n1;
    int e = j & 15, lane = (j >> 4) & 31, rest = j >> 9;
    int nt = rest & 63, ks = rest >> 6;
    int k = ks * 32 + b_k(lane, e);
    int col = nt * 16 + (lane & 15);
    fcwp[j] = (half_t)fcW[k * P_DIM + col];
  } else if (i < n3) {
    int j = i - n2;
    int u = j >> 6, k = j & 63;
    puh[j] = (half_t)uemb[(size_t)au[u] * H_DIM + k];
  } else if (i < n4) {
    int j = i - n3;
    poih[j] = (half_t)poi[j];
  }
}

// ---------------------------------------------------------------------------
// Kernel 1: GRU scan. 32 blocks x 128 threads; each block owns 16 users.
// Tile assignment nt = wid + 4*q  ->  q==2 is the n-gate tile for every wave
// (compile-time branch, no exec divergence). h lives in registers per thread;
// the gate phase writes new h directly into a 2KB f16 LDS buffer already in
// WMMA A-operand layout (one ds_store_b128 per thread), so the A-build is two
// aligned 32B LDS vector loads. x-embedding A operands are four aligned 16B
// f16 loads from the pre-converted poi slab (no converts on the scan's
// critical path).
// ---------------------------------------------------------------------------
__global__ void __launch_bounds__(128)
gru_kernel(const int* __restrict__ x, const float* __restrict__ h0,
           const half_t* __restrict__ poih,
           const half_t* __restrict__ wxp, const half_t* __restrict__ whp,
           const float* __restrict__ bias,
           half_t* __restrict__ outh, float* __restrict__ hT) {
  __shared__ alignas(16) half_t sAh[2 * 32 * 16];  // h in A-operand layout
  __shared__ float sG[16 * G3];
  __shared__ float sGn[16 * 64];
  int tid = threadIdx.x, lane = tid & 31, wid = tid >> 5;
  int ub = blockIdx.x * 16;

  // Per-thread ownership for the gate phase: row r, cols c0..c0+7.
  int r = tid >> 3;
  int c0 = (tid & 7) * 8;
  // Map (r, c0-block) -> contiguous 8-element run in A-layout LDS.
  int kstep = c0 >> 5;
  int k32 = c0 & 31;
  int lane2 = r + ((k32 & 8) ? 16 : 0);
  int ebase = (k32 & 16) ? 8 : 0;
  half_t* ah_dst = &sAh[(kstep * 32 + lane2) * 16 + ebase];

  float hreg[8];
  {
    h8pack p;
#pragma unroll
    for (int e2 = 0; e2 < 8; ++e2) {
      hreg[e2] = h0[(size_t)(ub + r) * 64 + c0 + e2];
      p.v[e2] = (half_t)hreg[e2];
    }
    *(h8pack*)ah_dst = p;
  }
  __syncthreads();

  const v16h* wxv = (const v16h*)wxp;
  const v16h* whv = (const v16h*)whp;
  int mrow = lane & 15;
  int u = ub + mrow;
  int rb = (lane & 16) ? 1 : 0;  // run base (units of 8 halfs) for this lane
  for (int tt = 0; tt < T_LEN; ++tt) {
    int xi = x[tt * U_LEN + u];
    const h8pack* prow = (const h8pack*)(poih + (size_t)xi * 64);
    h8pack r0 = prow[rb + 0], r1 = prow[rb + 2];  // kstep 0: K = kb..kb+7, kb+16..kb+23
    h8pack r2 = prow[rb + 4], r3 = prow[rb + 6];  // kstep 1
    v16h aX0, aX1;
#pragma unroll
    for (int e = 0; e < 8; ++e) {
      aX0[e] = r0.v[e];  aX0[e + 8] = r1.v[e];
      aX1[e] = r2.v[e];  aX1[e + 8] = r3.v[e];
    }
    v16h aH0 = *(const v16h*)&sAh[(0 * 32 + lane) * 16];
    v16h aH1 = *(const v16h*)&sAh[(1 * 32 + lane) * 16];
#pragma unroll
    for (int q = 0; q < 3; ++q) {
      int nt = wid + 4 * q;          // q==2 -> n-gate tile (uniform per wave)
      int col = nt * 16 + (lane & 15);
      v16h b0 = wxv[nt * 32 + lane];
      v16h b1 = wxv[(12 + nt) * 32 + lane];
      v16h c0v = whv[nt * 32 + lane];
      v16h c1v = whv[(12 + nt) * 32 + lane];
      float bb = bias[col];
      v8f accX = {bb, bb, bb, bb, bb, bb, bb, bb};
      accX = wmma16(aX0, b0, accX);
      accX = wmma16(aX1, b1, accX);
      if (q < 2) {                   // compile-time: r/z gates, fuse gx+gh
        accX = wmma16(aH0, c0v, accX);
        accX = wmma16(aH1, c1v, accX);
#pragma unroll
        for (int v = 0; v < 8; ++v)
          sG[(v + ((lane & 16) ? 8 : 0)) * G3 + col] = accX[v];
      } else {                       // compile-time: n gate, keep gh separate
        v8f accH = {};
        accH = wmma16(aH0, c0v, accH);
        accH = wmma16(aH1, c1v, accH);
#pragma unroll
        for (int v = 0; v < 8; ++v) {
          int rr = v + ((lane & 16) ? 8 : 0);
          sG[rr * G3 + col] = accX[v];
          sGn[rr * 64 + (col - 128)] = accH[v];
        }
      }
    }
    __syncthreads();
    h8pack p;
#pragma unroll
    for (int e2 = 0; e2 < 8; ++e2) {
      int c = c0 + e2;
      float rg = sigm(sG[r * G3 + c]);
      float zg = sigm(sG[r * G3 + 64 + c]);
      float ng = tanhf(sG[r * G3 + 128 + c] + rg * sGn[r * 64 + c]);
      float hn = (1.0f - zg) * hreg[e2] + zg * ng;
      hreg[e2] = hn;
      p.v[e2] = (half_t)hn;
      // transposed store: [u][h][t]
      outh[((size_t)(ub + r)) * H_DIM * T_LEN + (size_t)c * T_LEN + tt] = (half_t)hn;
      if (tt == T_LEN - 1) hT[(ub + r) * 64 + c] = hn;
    }
    *(h8pack*)ah_dst = p;
    __syncthreads();
  }
}

// ---------------------------------------------------------------------------
// Kernel 2: pairwise decay weights + causal weighted sum. 1 block / user,
// 4 waves split the 16 i-tiles. Weight tiles are generated directly in WMMA
// A-layout (each lane owns one row i); sum_w via an extra WMMA against an
// all-ones B (its D-layout matches the accumulators -> lane-local divide).
// The two exponentials are fused into one v_exp. B operands are direct
// aligned v16h loads from the [u][h][t] GRU output (L2-resident slab).
// ---------------------------------------------------------------------------
__global__ void __launch_bounds__(128)
pairwise_kernel(const float* __restrict__ tarr, const float* __restrict__ sarr,
                const half_t* __restrict__ outh, half_t* __restrict__ outw) {
  __shared__ float sT[T_LEN], sX[T_LEN], sY[T_LEN];
  int u = blockIdx.x;
  int tid = threadIdx.x, lane = tid & 31, wid = tid >> 5;
  for (int j = tid; j < T_LEN; j += 128) {
    sT[j] = tarr[j * U_LEN + u];
    sX[j] = sarr[((size_t)j * U_LEN + u) * 2 + 0];
    sY[j] = sarr[((size_t)j * U_LEN + u) * 2 + 1];
  }
  const v16h* outv = (const v16h*)(outh + (size_t)u * H_DIM * T_LEN);
  __builtin_prefetch(outv + lane, 0, 1);  // warm this user's 32KB slab
  __syncthreads();

  v16h ones;
#pragma unroll
  for (int e = 0; e < 16; ++e) ones[e] = (half_t)1.0f;
  const float W2PI = 7.2722052166430395e-5f;  // 2*pi/86400
  const float LTD  = 1.1574074074074074e-6f;  // 0.1/86400
  int hb = lane & 15;
  int khalf = (lane >> 4) & 1;  // lanes 16-31 take the upper 16 K rows

  for (int ti = wid; ti < 16; ti += 4) {
    int i = ti * 16 + (lane & 15);
    float it = sT[i], ix = sX[i], iy = sY[i];
    v8f acc0 = {}, acc1 = {}, acc2 = {}, acc3 = {}, accS = {};
    int nks = ti / 2 + 1;  // k-steps of 32 covering j <= ti*16+15
    for (int ks = 0; ks < nks; ++ks) {
      v16h aW;
#pragma unroll
      for (int e = 0; e < 16; ++e) {
        int j = ks * 32 + khalf * 16 + e;
        float w = 0.0f;
        if (j <= i) {
          float dt = it - sT[j];
          float dx = ix - sX[j], dy = iy - sY[j];
          float dsn = sqrtf(dx * dx + dy * dy);
          // exp(-dt*lamT/DAY)*exp(-ds*lamS) fused into one exponential
          float env = __expf(-(dt * LTD + dsn * 100.0f));
          w = 0.5f * (__cosf(dt * W2PI) + 1.0f) * env + 1e-10f;
        }
        aW[e] = (half_t)w;
      }
      accS = wmma16(aW, ones, accS);  // per-row sum of weights, broadcast over N
      // B: rows j = ks*32 + khalf*16 + e, col h = nt*16 + hb, contiguous in t
      int tv = ks * 2 + khalf;
      v16h b0 = outv[(0 * 16 + hb) * 16 + tv];
      v16h b1 = outv[(1 * 16 + hb) * 16 + tv];
      v16h b2 = outv[(2 * 16 + hb) * 16 + tv];
      v16h b3 = outv[(3 * 16 + hb) * 16 + tv];
      acc0 = wmma16(aW, b0, acc0);
      acc1 = wmma16(aW, b1, acc1);
      acc2 = wmma16(aW, b2, acc2);
      acc3 = wmma16(aW, b3, acc3);
    }
#pragma unroll
    for (int v = 0; v < 8; ++v) {
      int irow = ti * 16 + v + ((lane & 16) ? 8 : 0);
      float inv = 1.0f / accS[v];
      size_t base = ((size_t)irow * U_LEN + u) * 64 + (lane & 15);
      outw[base + 0]  = (half_t)(acc0[v] * inv);
      outw[base + 16] = (half_t)(acc1[v] * inv);
      outw[base + 32] = (half_t)(acc2[v] * inv);
      outw[base + 48] = (half_t)(acc3[v] * inv);
    }
  }
}

// ---------------------------------------------------------------------------
// Kernel 3: final FC GEMM [T*U,128] x [128,1024] + bias. One block per
// 16-row m-tile (one t, 16 users); 8 waves x 8 n-tiles; 4 chained WMMAs per
// 16x16 output tile; B operands are single v16h loads from the pre-swizzled
// fc weight slab (L2-resident). Store-bound (537 MB f32 output).
// ---------------------------------------------------------------------------
__global__ void __launch_bounds__(256)
fc_kernel(const half_t* __restrict__ outw, const half_t* __restrict__ puh,
          const half_t* __restrict__ fcwp, const float* __restrict__ fcb,
          float* __restrict__ yout) {
  __shared__ half_t sA[16 * 128];
  int mt = blockIdx.x;
  int tt = mt >> 5;
  int ub = (mt & 31) * 16;
  int tid = threadIdx.x, lane = tid & 31, wid = tid >> 5;
  const v16h* fcwv = (const v16h*)fcwp;
  __builtin_prefetch(fcwv + (size_t)(wid * 8) * 32 + lane, 0, 1);
  for (int i = tid; i < 16 * 128; i += 256) {
    int r = i >> 7, c = i & 127;
    sA[i] = (c < 64) ? outw[((size_t)tt * U_LEN + ub + r) * 64 + c]
                     : puh[(ub + r) * 64 + (c - 64)];
  }
  __syncthreads();

  v16h a0, a1, a2, a3;
#pragma unroll
  for (int e = 0; e < 16; ++e) {
    int base = (lane & 15) * 128;
    int k = a_k(lane, e);
    a0[e] = sA[base + k];
    a1[e] = sA[base + 32 + k];
    a2[e] = sA[base + 64 + k];
    a3[e] = sA[base + 96 + k];
  }
#pragma unroll 1
  for (int q = 0; q < 8; ++q) {
    int nt = wid * 8 + q;
    int col = nt * 16 + (lane & 15);
    v16h b0 = fcwv[(size_t)(0 * 64 + nt) * 32 + lane];
    v16h b1 = fcwv[(size_t)(1 * 64 + nt) * 32 + lane];
    v16h b2 = fcwv[(size_t)(2 * 64 + nt) * 32 + lane];
    v16h b3 = fcwv[(size_t)(3 * 64 + nt) * 32 + lane];
    float bb = fcb[col];
    v8f acc = {bb, bb, bb, bb, bb, bb, bb, bb};
    acc = wmma16(a0, b0, acc);
    acc = wmma16(a1, b1, acc);
    acc = wmma16(a2, b2, acc);
    acc = wmma16(a3, b3, acc);
#pragma unroll
    for (int v = 0; v < 8; ++v) {
      int m = tt * U_LEN + ub + v + ((lane & 16) ? 8 : 0);
      yout[(size_t)m * P_DIM + col] = acc[v];
    }
  }
}

// ---------------------------------------------------------------------------
extern "C" void kernel_launch(void* const* d_in, const int* in_sizes, int n_in,
                              void* d_out, int out_size, void* d_ws, size_t ws_size,
                              hipStream_t stream) {
  (void)in_sizes; (void)n_in; (void)out_size; (void)ws_size;
  const int*   x    = (const int*)d_in[0];
  const float* tarr = (const float*)d_in[1];
  // d_in[2] t_slot unused by reference
  const float* sarr = (const float*)d_in[3];
  // d_in[4] y unused by reference
  const float* h0   = (const float*)d_in[5];
  const int*   au   = (const int*)d_in[6];
  const float* poi  = (const float*)d_in[7];
  const float* uemb = (const float*)d_in[8];
  const float* Wx   = (const float*)d_in[9];
  const float* Wh   = (const float*)d_in[10];
  const float* bias = (const float*)d_in[11];
  const float* fcW  = (const float*)d_in[12];
  const float* fcb  = (const float*)d_in[13];

  float* yout = (float*)d_out;
  float* hT   = yout + (size_t)T_LEN * U_LEN * P_DIM;

  half_t* wxp  = (half_t*)d_ws;                       // B-swizzled Wx
  half_t* whp  = wxp + 64 * G3;                       // B-swizzled Wh
  half_t* fcwp = whp + 64 * G3;                       // B-swizzled fc_W
  half_t* puh  = fcwp + 128 * P_DIM;                  // gathered p_u (f16)
  half_t* poih = puh + U_LEN * H_DIM;                 // poi_emb f16 copy
  half_t* outh = poih + P_DIM * H_DIM;                // GRU out, [u][h][t] f16
  half_t* outw = outh + (size_t)U_LEN * H_DIM * T_LEN;// weighted out, [t][u][h] f16

  int prepN = 64 * G3 * 2 + 128 * P_DIM + U_LEN * H_DIM + P_DIM * H_DIM;
  prep_kernel<<<(prepN + 255) / 256, 256, 0, stream>>>(Wx, Wh, fcW, au, uemb, poi,
                                                       wxp, whp, fcwp, puh, poih);
  gru_kernel<<<32, 128, 0, stream>>>(x, h0, poih, wxp, whp, bias, outh, hT);
  pairwise_kernel<<<512, 128, 0, stream>>>(tarr, sarr, outh, outw);
  fc_kernel<<<8192, 256, 0, stream>>>(outw, puh, fcwp, fcb, yout);
}
